// TransCorrBlock_70609262346370
// MI455X (gfx1250) — matrix operand order repaired and checked by reading, not compiled, not run on hardware
//
#include <hip/hip_runtime.h>

// ---------------- types ----------------
typedef __bf16 bf16;
typedef __attribute__((ext_vector_type(8)))  bf16  v8bf;
typedef __attribute__((ext_vector_type(16))) bf16  v16bf;
typedef __attribute__((ext_vector_type(8)))  float v8f;

// Problem constants: B=2, C=256, H=W=64, D=256, L=4, R=4, P=9
#define kHW   4096
#define kN    8192     // B*HW
#define kC    256
#define kD    256

// ---------------- WMMA fragment loaders (ISA 7.12.2 layouts) ----------------
// A-matrix 16x32 bf16: lane<16 holds row M=lane, K={0..7,16..23}; lane>=16 row M=lane-16, K={8..15,24..31}
__device__ __forceinline__ v16bf frag_a(const bf16* base, int ld, int lane) {
  const int r   = lane & 15;
  const int off = (lane >> 4) << 3;              // 0 or 8
  const bf16* p = base + (size_t)r * ld + off;
  v8bf lo = *(const v8bf*)(p);
  v8bf hi = *(const v8bf*)(p + 16);
  return __builtin_shufflevector(lo, hi, 0,1,2,3,4,5,6,7,8,9,10,11,12,13,14,15);
}
// B-matrix 32x16 bf16 (loaded from B^T row-major): lane<16 holds col N=lane, K={0..7,16..23}; lane>=16 K={8..15,24..31}
__device__ __forceinline__ v16bf frag_b(const bf16* baseT, int ld, int lane) {
  const int c   = lane & 15;
  const int off = (lane >> 4) << 3;              // 0 or 8
  const bf16* p = baseT + (size_t)c * ld + off;
  v8bf lo = *(const v8bf*)(p);
  v8bf hi = *(const v8bf*)(p + 16);
  return __builtin_shufflevector(lo, hi, 0,1,2,3,4,5,6,7,8,9,10,11,12,13,14,15);
}

// 32B global -> LDS, no VGPR staging (ASYNCcnt-tracked). offset:16 advances both addresses.
__device__ __forceinline__ void async_row32(const bf16* grow, unsigned lds_off) {
  unsigned long long ga = (unsigned long long)(size_t)grow;
  asm volatile("global_load_async_to_lds_b128 %0, %1, off"
               :: "v"(lds_off), "v"(ga) : "memory");
  asm volatile("global_load_async_to_lds_b128 %0, %1, off offset:16"
               :: "v"(lds_off), "v"(ga) : "memory");
}

// ---------------- prep: weight transpose fp32 [K][N] -> bf16 [N][K] ----------------
__global__ void k_transW(const float* __restrict__ Wsrc, bf16* __restrict__ WT) {
  int t = blockIdx.x * 256 + threadIdx.x;        // 65536 total
  int n = t >> 8, k = t & 255;
  WT[t] = (bf16)Wsrc[k * 256 + n];
}

// ---------------- prep: fmap fp32 [B,C,HW] -> bf16 [B*HW, C] (LDS tile transpose) ----
__global__ void k_featT(const float* __restrict__ fmap, bf16* __restrict__ outT) {
  __shared__ float tile[32][33];
  int b   = blockIdx.z;
  int hw0 = blockIdx.x * 32;
  int c0  = blockIdx.y * 32;
  int tx = threadIdx.x, ty = threadIdx.y;
  tile[ty][tx] = fmap[((size_t)(b * kC + c0 + ty)) * kHW + hw0 + tx];
  __syncthreads();
  outT[((size_t)(b * kHW + hw0 + ty)) * kC + c0 + tx] = (bf16)tile[tx][ty];
}

// ---------------- encoder GEMM: x = featT @ Wf + bf + bp + pos@Wp (fp32 out) -------
// block = 128 threads (4 waves), tile = 16 rows x 256 cols; wave w covers cols [64w,64w+64)
__global__ void k_encode(const bf16* __restrict__ featT, const bf16* __restrict__ WfT,
                         const float* __restrict__ bfv, const float* __restrict__ bpv,
                         const float* __restrict__ Wp, const float* __restrict__ coords,
                         int mode, float* __restrict__ xout) {
  const int m0   = blockIdx.x * 16;
  const int wid  = threadIdx.x >> 5;
  const int lane = threadIdx.x & 31;
  const int n0   = wid * 64;
  v8f acc[4] = {};
  for (int kk = 0; kk < kC; kk += 32) {
    v16bf a = frag_a(featT + (size_t)m0 * kC + kk, kC, lane);
#pragma unroll
    for (int nf = 0; nf < 4; nf++) {
      v16bf b = frag_b(WfT + (size_t)(n0 + nf * 16) * kC + kk, kC, lane);
      acc[nf] = __builtin_amdgcn_wmma_f32_16x16x32_bf16(false, a, false, b,
                                                        (short)0, acc[nf], false, false);
    }
  }
  const int hi = lane >> 4, lo = lane & 15;
#pragma unroll
  for (int nf = 0; nf < 4; nf++) {
    int d = n0 + nf * 16 + lo;
    float bias = bfv[d] + bpv[d];
    float wp0 = Wp[d], wp1 = Wp[kD + d];
#pragma unroll
    for (int r = 0; r < 8; r++) {
      int m  = m0 + r + (hi << 3);
      int b  = m >> 12, hw = m & (kHW - 1);
      float xx, yy;
      if (mode == 1) {            // frame1: coords1 channels are (x,y); pos uses (y,x)
        xx = coords[((b * 2 + 0) << 12) + hw];
        yy = coords[((b * 2 + 1) << 12) + hw];
      } else {                    // frame2: integer grid
        yy = (float)(hw >> 6);
        xx = (float)(hw & 63);
      }
      xout[(size_t)m * kD + d] = acc[nf][r] + bias + yy * wp0 + xx * wp1;
    }
  }
}

// ---------------- LayerNorm over D=256 (one block per row) -> bf16 ----------------
__global__ void k_layernorm(const float* __restrict__ xin, const float* __restrict__ g,
                            const float* __restrict__ bb, bf16* __restrict__ vout) {
  __shared__ float red[256];
  int row = blockIdx.x, d = threadIdx.x;
  float x = xin[(size_t)row * kD + d];
  red[d] = x; __syncthreads();
  for (int s = 128; s > 0; s >>= 1) { if (d < s) red[d] += red[d + s]; __syncthreads(); }
  float mu = red[0] * (1.0f / 256.0f);
  __syncthreads();
  float xc = x - mu;
  red[d] = xc * xc; __syncthreads();
  for (int s = 128; s > 0; s >>= 1) { if (d < s) red[d] += red[d + s]; __syncthreads(); }
  float var = red[0] * (1.0f / 256.0f);
  vout[(size_t)row * kD + d] = (bf16)(g[d] * xc * rsqrtf(var + 1e-5f) + bb[d]);
}

// ---------------- projection GEMM: out = (A @ W) * scale -> bf16 -------------------
__global__ void k_proj(const bf16* __restrict__ A, const bf16* __restrict__ WT,
                       float scale, bf16* __restrict__ outp) {
  const int m0   = blockIdx.x * 16;
  const int wid  = threadIdx.x >> 5;
  const int lane = threadIdx.x & 31;
  const int n0   = wid * 64;
  v8f acc[4] = {};
  for (int kk = 0; kk < kD; kk += 32) {
    v16bf a = frag_a(A + (size_t)m0 * kD + kk, kD, lane);
#pragma unroll
    for (int nf = 0; nf < 4; nf++) {
      v16bf b = frag_b(WT + (size_t)(n0 + nf * 16) * kD + kk, kD, lane);
      acc[nf] = __builtin_amdgcn_wmma_f32_16x16x32_bf16(false, a, false, b,
                                                        (short)0, acc[nf], false, false);
    }
  }
  const int hi = lane >> 4, lo = lane & 15;
#pragma unroll
  for (int nf = 0; nf < 4; nf++) {
    int d = n0 + nf * 16 + lo;
#pragma unroll
    for (int r = 0; r < 8; r++) {
      int m = m0 + r + (hi << 3);
      outp[(size_t)m * kD + d] = (bf16)(acc[nf][r] * scale);
    }
  }
}

// ---------------- corr GEMM: corr[b][n][m] = q[b][n]·k[b][m]  (A @ B^T) ------------
// block 256 threads (8 waves as 2x4), tile 128x128.
// Double-buffered LDS fed by GLOBAL_LOAD_ASYNC_TO_LDS_B128 (ASYNCcnt pipeline):
//   per K-step: wait tile k, barrier, read fragments(k), issue async tile k+1, 8x WMMA.
// Buffer safety: buffer written by issue(k) was last read at frags(k-1); barrier(k) separates them.
__global__ void k_corr(const bf16* __restrict__ qb, const bf16* __restrict__ kb,
                       bf16* __restrict__ corr0) {
  __shared__ __align__(16) bf16 lA[2][128 * 40];
  __shared__ __align__(16) bf16 lB[2][128 * 40];
  const int b  = blockIdx.z;
  const int n0 = blockIdx.x * 128;   // columns (k rows)
  const int m0 = blockIdx.y * 128;   // rows (q rows)
  const bf16* qbase = qb + ((size_t)b * kHW + m0) * kD;
  const bf16* kbase = kb + ((size_t)b * kHW + n0) * kD;
  const int tid = threadIdx.x;
  const int lane = tid & 31, wid = tid >> 5;
  const int wm = wid & 1, wn = wid >> 1;         // 2 x 4 wave grid
  const int lrow = tid >> 1, lhalf = tid & 1;    // each thread moves one 32B row-segment
  const unsigned slot = (unsigned)((lrow * 40 + lhalf * 16) * 2);
  const unsigned baseA[2] = { (unsigned)(size_t)&lA[0][0], (unsigned)(size_t)&lA[1][0] };
  const unsigned baseB[2] = { (unsigned)(size_t)&lB[0][0], (unsigned)(size_t)&lB[1][0] };

  v8f acc[4][2] = {};

  // prologue: tile 0 -> buffer 0
  async_row32(qbase + (size_t)lrow * kD + 0 + lhalf * 16, baseA[0] + slot);
  async_row32(kbase + (size_t)lrow * kD + 0 + lhalf * 16, baseB[0] + slot);

  for (int kk = 0; kk < kD; kk += 32) {
    const int cur = (kk >> 5) & 1;
    asm volatile("s_wait_asynccnt 0x0" ::: "memory");
    __syncthreads();                                   // tile kk visible to all waves
    v16bf af[4], bfr[2];
#pragma unroll
    for (int mf = 0; mf < 4; mf++) af[mf]  = frag_a(&lA[cur][(wm * 64 + mf * 16) * 40], 40, lane);
#pragma unroll
    for (int nf = 0; nf < 2; nf++) bfr[nf] = frag_b(&lB[cur][(wn * 32 + nf * 16) * 40], 40, lane);
    if (kk + 32 < kD) {                                // overlap next copy with WMMA
      const int nxt = cur ^ 1;
      async_row32(qbase + (size_t)lrow * kD + (kk + 32) + lhalf * 16, baseA[nxt] + slot);
      async_row32(kbase + (size_t)lrow * kD + (kk + 32) + lhalf * 16, baseB[nxt] + slot);
    }
#pragma unroll
    for (int mf = 0; mf < 4; mf++)
#pragma unroll
      for (int nf = 0; nf < 2; nf++)
        acc[mf][nf] = __builtin_amdgcn_wmma_f32_16x16x32_bf16(false, af[mf], false, bfr[nf],
                                                              (short)0, acc[mf][nf], false, false);
  }
  const int hi = lane >> 4, lo = lane & 15;
#pragma unroll
  for (int mf = 0; mf < 4; mf++)
#pragma unroll
    for (int nf = 0; nf < 2; nf++)
#pragma unroll
      for (int r = 0; r < 8; r++) {
        int row = m0 + wm * 64 + mf * 16 + r + (hi << 3);
        int col = n0 + wn * 32 + nf * 16 + lo;
        corr0[((size_t)b * kHW + row) * kHW + col] = (bf16)acc[mf][nf][r];
      }
}

// ---------------- 2x2 average pool (bf16 in/out, fp32 accumulate) ------------------
__global__ void k_pool(const bf16* __restrict__ in, bf16* __restrict__ out, int Hi) {
  int Ho = Hi >> 1, Wo = Hi >> 1, Wi = Hi;
  size_t total = (size_t)kN * Ho * Wo;
  size_t t = (size_t)blockIdx.x * blockDim.x + threadIdx.x;
  if (t >= total) return;
  int x = (int)(t % Wo);
  size_t r = t / Wo;
  int y = (int)(r % Ho);
  size_t img = r / Ho;
  const bf16* ip = in + img * (size_t)(Hi * Wi);
  float s = (float)ip[(2 * y) * Wi + 2 * x]     + (float)ip[(2 * y) * Wi + 2 * x + 1]
          + (float)ip[(2 * y + 1) * Wi + 2 * x] + (float)ip[(2 * y + 1) * Wi + 2 * x + 1];
  out[t] = (bf16)(0.25f * s);
}

// ---------------- multi-level windowed bilinear sampling ---------------------------
__device__ __forceinline__ float bsample(const bf16* img, int Hl, int Wl, float yy, float xx) {
  bool valid = (xx >= 0.0f) && (xx <= (float)(Wl - 1)) && (yy >= 0.0f) && (yy <= (float)(Hl - 1));
  int xi = (int)fminf(fmaxf(xx, 0.0f), (float)(Wl - 1));
  int yi = (int)fminf(fmaxf(yy, 0.0f), (float)(Hl - 1));
  float v = (float)img[yi * Wl + xi];
  return valid ? v : 0.0f;
}

__global__ void k_sample(const float* __restrict__ coords_q,
                         const bf16* __restrict__ c0, const bf16* __restrict__ c1,
                         const bf16* __restrict__ c2, const bf16* __restrict__ c3,
                         float* __restrict__ out) {
  int t = blockIdx.x * 256 + threadIdx.x;
  if (t >= 2 * 324 * kHW) return;
  int hw   = t & (kHW - 1);
  int rest = t >> 12;                 // b*324 + i*81 + j
  int b = rest / 324;
  int c = rest - b * 324;
  int i = c / 81;
  int j = c - i * 81;
  float cx = coords_q[((b * 2 + 0) << 12) + hw];
  float cy = coords_q[((b * 2 + 1) << 12) + hw];
  float inv = 1.0f / (float)(1 << i);
  float dx = (float)(j / 9 - 4);      // delta[...,0] adds to x
  float dy = (float)(j % 9 - 4);
  float x = cx * inv + dx, y = cy * inv + dy;
  int Hl = 64 >> i, Wl = Hl;
  const bf16* img;
  if (i == 0) img = c0; else if (i == 1) img = c1; else if (i == 2) img = c2; else img = c3;
  img += ((size_t)(b * kHW + hw)) * (size_t)(Hl * Wl);
  float x0 = floorf(x), y0 = floorf(y);
  float wx = x - x0, wy = y - y0;
  float v00 = bsample(img, Hl, Wl, y0,        x0);
  float v01 = bsample(img, Hl, Wl, y0,        x0 + 1.0f);
  float v10 = bsample(img, Hl, Wl, y0 + 1.0f, x0);
  float v11 = bsample(img, Hl, Wl, y0 + 1.0f, x0 + 1.0f);
  out[t] = (1.0f - wy) * (1.0f - wx) * v00 + (1.0f - wy) * wx * v01
         + wy * (1.0f - wx) * v10 + wy * wx * v11;
}

// ---------------- host-side orchestration -----------------------------------------
extern "C" void kernel_launch(void* const* d_in, const int* in_sizes, int n_in,
                              void* d_out, int out_size, void* d_ws, size_t ws_size,
                              hipStream_t stream) {
  (void)in_sizes; (void)n_in; (void)out_size; (void)ws_size;
  const float* fmap1   = (const float*)d_in[0];
  const float* fmap2   = (const float*)d_in[1];
  const float* coords1 = (const float*)d_in[2];
  const float* coordsq = (const float*)d_in[3];
  const float* Wf      = (const float*)d_in[4];
  const float* bfv     = (const float*)d_in[5];
  const float* Wp      = (const float*)d_in[6];
  const float* bpv     = (const float*)d_in[7];
  const float* ln_g    = (const float*)d_in[8];
  const float* ln_b    = (const float*)d_in[9];
  const float* Wq      = (const float*)d_in[10];
  const float* Wk      = (const float*)d_in[11];
  float* out = (float*)d_out;

  char* ws = (char*)d_ws;
  size_t off = 0;
  auto take = [&](size_t bytes) -> char* {
    char* p = ws + off;
    off = (off + bytes + 255) & ~(size_t)255;
    return p;
  };
  bf16*  WfT = (bf16*)take((size_t)kC * kD * 2);
  bf16*  WqT = (bf16*)take((size_t)kD * kD * 2);
  bf16*  WkT = (bf16*)take((size_t)kD * kD * 2);
  bf16*  f1T = (bf16*)take((size_t)kN * kC * 2);
  bf16*  f2T = (bf16*)take((size_t)kN * kC * 2);
  float* x1  = (float*)take((size_t)kN * kD * 4);
  float* x2  = (float*)take((size_t)kN * kD * 4);
  bf16*  v1b = (bf16*)take((size_t)kN * kD * 2);
  bf16*  v2b = (bf16*)take((size_t)kN * kD * 2);
  bf16*  qb  = (bf16*)take((size_t)kN * kD * 2);
  bf16*  kb  = (bf16*)take((size_t)kN * kD * 2);
  bf16*  c0  = (bf16*)take((size_t)kN * 4096 * 2);
  bf16*  c1  = (bf16*)take((size_t)kN * 1024 * 2);
  bf16*  c2  = (bf16*)take((size_t)kN * 256 * 2);
  bf16*  c3  = (bf16*)take((size_t)kN * 64 * 2);

  // weights -> bf16 transposed
  k_transW<<<256, 256, 0, stream>>>(Wf, WfT);
  k_transW<<<256, 256, 0, stream>>>(Wq, WqT);
  k_transW<<<256, 256, 0, stream>>>(Wk, WkT);
  // fmaps -> bf16 [N][C]
  dim3 gT(128, 8, 2), bT(32, 32);
  k_featT<<<gT, bT, 0, stream>>>(fmap1, f1T);
  k_featT<<<gT, bT, 0, stream>>>(fmap2, f2T);
  // encode + LN
  k_encode<<<kN / 16, 128, 0, stream>>>(f1T, WfT, bfv, bpv, Wp, coords1, 1, x1);
  k_encode<<<kN / 16, 128, 0, stream>>>(f2T, WfT, bfv, bpv, Wp, coords1, 2, x2);
  k_layernorm<<<kN, 256, 0, stream>>>(x1, ln_g, ln_b, v1b);
  k_layernorm<<<kN, 256, 0, stream>>>(x2, ln_g, ln_b, v2b);
  // q (scaled by 1/sqrt(D)=1/16), k
  k_proj<<<kN / 16, 128, 0, stream>>>(v1b, WqT, 1.0f / 16.0f, qb);
  k_proj<<<kN / 16, 128, 0, stream>>>(v2b, WkT, 1.0f, kb);
  // corr = q k^T  (async-LDS double-buffered WMMA)
  dim3 gC(32, 32, 2);
  k_corr<<<gC, 256, 0, stream>>>(qb, kb, c0);
  // pyramid
  k_pool<<<32768, 256, 0, stream>>>(c0, c1, 64);
  k_pool<<<8192, 256, 0, stream>>>(c1, c2, 32);
  k_pool<<<2048, 256, 0, stream>>>(c2, c3, 16);
  // sampling -> fp32 output [B, 324, H, W]
  k_sample<<<(2 * 324 * kHW + 255) / 256, 256, 0, stream>>>(coordsq, c0, c1, c2, c3, out);
}
